// XNORFactorizedReduce_19138374271268
// MI455X (gfx1250) — 8-bit matrix operand order repaired and checked by reading, compiled, not run on hardware
//
#include <hip/hip_runtime.h>
#include <hip/hip_bf16.h>
#include <stddef.h>

typedef __attribute__((ext_vector_type(8))) int v8i;

#define CH   192
#define HH   64
#define WW   64
#define HP   66
#define BB   32
#define HO   32
#define WO   32
#define KTAP 3

__global__ void stats_kernel(const float* __restrict__ x,
                             const float* __restrict__ g1, const float* __restrict__ b1,
                             const float* __restrict__ g2, const float* __restrict__ b2,
                             float* __restrict__ coef)
{
    int c = blockIdx.x;
    int tid = threadIdx.x;
    float sum = 0.f, sumsq = 0.f;
    for (int b = 0; b < BB; ++b) {
        const float* p = x + ((size_t)(b * CH + c)) * (HH * WW);
        for (int i = tid; i < HH * WW; i += 256) {
            float v = p[i];
            sum += v; sumsq += v * v;
        }
    }
    __shared__ float sA[256], sB[256];
    sA[tid] = sum; sB[tid] = sumsq;
    __syncthreads();
    for (int off = 128; off > 0; off >>= 1) {
        if (tid < off) { sA[tid] += sA[tid + off]; sB[tid] += sB[tid + off]; }
        __syncthreads();
    }
    if (tid == 0) {
        const float n = (float)(BB * HH * WW);
        float mu  = sA[0] / n;
        float var = sB[0] / n - mu * mu;
        float rs  = rsqrtf(var + 1e-5f);
        float s1 = g1[c] * rs, t1 = b1[c] - s1 * mu;
        float s2 = g2[c] * rs, t2 = b2[c] - s2 * mu;
        coef[c]           = s1;
        coef[CH + c]      = t1;
        coef[2 * CH + c]  = s2;
        coef[3 * CH + c]  = t2;
    }
}

__global__ void halo_zero_kernel(signed char* __restrict__ a1,
                                 signed char* __restrict__ a2)
{
    int idx = blockIdx.x * 256 + threadIdx.x;
    if (idx >= BB * 260 * 12) return;
    int chunk = idx % 12;
    int p = idx / 12;
    int b = p / 260;
    int q = p % 260;
    int h, w;
    if (q < 66)       { h = 0;  w = q; }
    else if (q < 132) { h = 65; w = q - 66; }
    else { int s = q - 132; h = 1 + (s >> 1); w = (s & 1) * 65; }
    size_t off = (((size_t)b * HP + h) * HP + w) * CH + chunk * 16;
    *(int4*)(a1 + off) = make_int4(0, 0, 0, 0);
    *(int4*)(a2 + off) = make_int4(0, 0, 0, 0);
}

__global__ void binarize_kernel(const float* __restrict__ x,
                                const float* __restrict__ coef,
                                signed char* __restrict__ act1,
                                signed char* __restrict__ act2)
{
    int bh = blockIdx.x;
    int b = bh >> 6, h = bh & 63;
    int tid = threadIdx.x;
    __shared__ __align__(16) signed char l1[WW * CH];
    __shared__ __align__(16) signed char l2[WW * CH];

    for (int idx = tid; idx < WW * CH; idx += 256) {
        int c = idx >> 6, w = idx & 63;
        float v  = x[(((size_t)b * CH + c) * HH + h) * WW + w];
        float h1 = coef[c] * v + coef[CH + c];
        float h2 = coef[2 * CH + c] * v + coef[3 * CH + c];
        l1[w * CH + c] = (signed char)((h1 > 0.f) - (h1 < 0.f));
        l2[w * CH + c] = (signed char)((h2 > 0.f) - (h2 < 0.f));
    }
    __syncthreads();
    size_t base = (((size_t)b * HP + (h + 1)) * HP + 1) * CH;
    const int4* p1 = (const int4*)l1;
    const int4* p2 = (const int4*)l2;
    int4* o1 = (int4*)(act1 + base);
    int4* o2 = (int4*)(act2 + base);
    for (int idx = tid; idx < (WW * CH) / 16; idx += 256) {
        o1[idx] = p1[idx];
        o2[idx] = p2[idx];
    }
}

__global__ void wpack_kernel(const float* __restrict__ w,
                             unsigned int* __restrict__ wp)
{
    int idx = blockIdx.x * 256 + threadIdx.x;
    if (idx >= 12 * 27 * 32 * 8) return;
    int v     = idx & 7;
    int lane  = (idx >> 3) & 31;
    int grp   = idx >> 8;
    int kc    = grp % 27;
    int ntile = grp / 27;
    int co    = ntile * 16 + (lane & 15);
    int half  = lane >> 4;
    int tap   = kc / 3;
    int kh = tap / 3, kw = tap - kh * 3;
    int cbase = (kc % 3) * 64;
    int kword = (v & 1) + half * 2 + (v >> 1) * 4;
    int ci0   = cbase + kword * 4;
    unsigned int word = 0;
    for (int j = 0; j < 4; ++j) {
        float wv = w[(((size_t)co * CH + (ci0 + j)) * KTAP + kh) * KTAP + kw];
        int s = (wv > 0.f) - (wv < 0.f);
        word |= ((unsigned int)(unsigned char)(signed char)s) << (8 * j);
    }
    wp[idx] = word;
}

__global__ void __launch_bounds__(128) xnor_gemm_kernel(
    const signed char* __restrict__ act1, const signed char* __restrict__ act2,
    const unsigned int* __restrict__ wp1, const unsigned int* __restrict__ wp2,
    const float* __restrict__ bias1, const float* __restrict__ bias2,
    const float* __restrict__ x,
    float* __restrict__ out)
{
    int wib  = threadIdx.x >> 5;
    int wave = blockIdx.x * 4 + wib;
    int lane = threadIdx.x & 31;
    int branch = (wave >= 24576) ? 1 : 0;
    int t = wave - branch * 24576;
    int ntile = t % 12;
    int mtile = t / 12;
    int b   = mtile >> 6;
    int rem = mtile & 63;
    int oh  = rem >> 1;
    int ow0 = (rem & 1) << 4;
    int r    = lane & 15;
    int half = lane >> 4;
    int ow   = ow0 + r;

    const signed char*  act  = branch ? act2  : act1;
    const unsigned int* wp   = branch ? wp2   : wp1;
    const float*        bias = branch ? bias2 : bias1;

    const unsigned long long* ap = (const unsigned long long*)
        (act + (((size_t)b * HP + 2 * oh) * HP + 2 * ow) * CH + half * 8);
    const uint4* bq = (const uint4*)wp + (size_t)ntile * 27 * 64 + lane * 2;

    v8i acc0 = {};
    v8i acc1 = {};

#pragma unroll
    for (int kh = 0; kh < 3; ++kh) {
#pragma unroll
        for (int kwp = 0; kwp < 3; ++kwp) {
#pragma unroll
            for (int cg = 0; cg < 3; ++cg) {
                const int kc = (kh * 3 + kwp) * 3 + cg;
                union { v8i v; unsigned long long u[4]; } A;
#pragma unroll
                for (int g = 0; g < 4; ++g)
                    A.u[g] = ap[kh * 1584 + kwp * 24 + cg * 8 + g * 2];
                union { v8i v; uint4 q[2]; } Bm;
                Bm.q[0] = bq[kc * 64];
                Bm.q[1] = bq[kc * 64 + 1];
                if (kc & 1)
                    acc1 = __builtin_amdgcn_wmma_i32_16x16x64_iu8(
                        true, A.v, true, Bm.v, acc1, false, false);
                else
                    acc0 = __builtin_amdgcn_wmma_i32_16x16x64_iu8(
                        true, A.v, true, Bm.v, acc0, false, false);
            }
        }
    }

    __shared__ __align__(16) float lt[4][16][17];
    int co = ntile * 16 + r;
    float bv = bias[co];
#pragma unroll
    for (int v = 0; v < 8; ++v) {
        float val = (float)(acc0[v] + acc1[v]) + bv;
        lt[wib][v + 8 * half][r] = val > 0.f ? val : 0.f;
    }
    __syncthreads();

    int co_r = lane >> 1;
    int part = lane & 1;
    int co2  = ntile * 16 + co_r;
    int owb  = ow0 + part * 8;
    size_t ob = (((size_t)b * (2 * CH) + branch * CH + co2) * HO + oh) * WO + owb;
    const float* px = x + (((size_t)b * CH + co2) * HH + 2 * oh) * WW + 2 * owb;
    float res[8];
#pragma unroll
    for (int j = 0; j < 8; ++j) {
        float sc = 0.25f * (px[2 * j] + px[2 * j + 1] + px[2 * j + WW] + px[2 * j + WW + 1]);
        res[j] = lt[wib][part * 8 + j][co_r] + sc;
    }
    *(float4*)(out + ob)     = make_float4(res[0], res[1], res[2], res[3]);
    *(float4*)(out + ob + 4) = make_float4(res[4], res[5], res[6], res[7]);
}

extern "C" void kernel_launch(void* const* d_in, const int* in_sizes, int n_in,
                              void* d_out, int out_size, void* d_ws, size_t ws_size,
                              hipStream_t stream)
{
    const float* x     = (const float*)d_in[0];
    const float* g1    = (const float*)d_in[1];
    const float* b1    = (const float*)d_in[2];
    const float* w1    = (const float*)d_in[3];
    const float* bias1 = (const float*)d_in[4];
    const float* g2    = (const float*)d_in[5];
    const float* b2    = (const float*)d_in[6];
    const float* w2    = (const float*)d_in[7];
    const float* bias2 = (const float*)d_in[8];
    float* out = (float*)d_out;

    const size_t actBytes = (size_t)BB * HP * HP * CH;
    const size_t wpWords  = (size_t)12 * 27 * 32 * 8;
    char* ws = (char*)d_ws;
    float*        coef = (float*)ws;
    signed char*  act1 = (signed char*)(ws + 4096);
    signed char*  act2 = (signed char*)(ws + 4096 + actBytes);
    unsigned int* wpk1 = (unsigned int*)(ws + 4096 + 2 * actBytes);
    unsigned int* wpk2 = wpk1 + wpWords;

    stats_kernel<<<CH, 256, 0, stream>>>(x, g1, b1, g2, b2, coef);
    halo_zero_kernel<<<(BB * 260 * 12 + 255) / 256, 256, 0, stream>>>(act1, act2);
    binarize_kernel<<<BB * HH, 256, 0, stream>>>(x, coef, act1, act2);
    wpack_kernel<<<(int)((wpWords + 255) / 256), 256, 0, stream>>>(w1, wpk1);
    wpack_kernel<<<(int)((wpWords + 255) / 256), 256, 0, stream>>>(w2, wpk2);
    xnor_gemm_kernel<<<12288, 128, 0, stream>>>(act1, act2, wpk1, wpk2,
                                                bias1, bias2, x, out);
}
